// GraphAttentionLayer_62732292326127
// MI455X (gfx1250) — matrix operand order repaired and checked by reading, not compiled, not run on hardware
//
#include <hip/hip_runtime.h>

#define GAT_N     8192
#define GAT_INF   256
#define GAT_OUTF  128
#define GAT_ALPHA 0.2f

typedef __attribute__((ext_vector_type(8)))  float  v8f;
typedef __attribute__((ext_vector_type(8)))  __bf16 v8bf;
typedef __attribute__((ext_vector_type(16))) __bf16 v16bf;

__device__ __forceinline__ __bf16 f2bf(float x) { return (__bf16)x; }

// -------------------------------------------------------------------------
// Kernel 1: Wh = h @ W  via v_wmma_f32_16x16x32_bf16.
// One wave per 16x16 output tile. Outputs are stored TRANSPOSED
// (feature-major: X[n][row]) so that (a) the D fragment's 8 rows per lane
// are contiguous -> b128 stores, and (b) the attention kernel's B fragments
// become contiguous 32-byte loads.
// -------------------------------------------------------------------------
__global__ void gat_wh_kernel(const float* __restrict__ h,
                              const float* __restrict__ W,
                              float* __restrict__ WhfT,    // [OUT_F][N] f32
                              __bf16* __restrict__ WhbT)   // [OUT_F][N] bf16
{
    const int lane = threadIdx.x & 31;
    const int wave = threadIdx.x >> 5;
    const int half = lane >> 4;      // K-group select per ISA A/B layouts
    const int m    = lane & 15;
    const int tile = blockIdx.x * 8 + wave;     // 4096 tiles total
    const int r0   = (tile >> 3) * 16;          // 512 row tiles
    const int n0   = (tile & 7) * 16;           // 8 col tiles

    v8f acc = {};
    for (int kt = 0; kt < GAT_INF; kt += 32) {
        v16bf A, B;
        // A (16x32 bf16): lane m = row; K = half*8 + {0..7} and {16..23}
        const float4* hp = (const float4*)(h + (size_t)(r0 + m) * GAT_INF + kt + half * 8);
        const float4 h0 = hp[0], h1 = hp[1], h2 = hp[4], h3 = hp[5];
        A[0]=f2bf(h0.x); A[1]=f2bf(h0.y); A[2]=f2bf(h0.z); A[3]=f2bf(h0.w);
        A[4]=f2bf(h1.x); A[5]=f2bf(h1.y); A[6]=f2bf(h1.z); A[7]=f2bf(h1.w);
        A[8]=f2bf(h2.x); A[9]=f2bf(h2.y); A[10]=f2bf(h2.z); A[11]=f2bf(h2.w);
        A[12]=f2bf(h3.x); A[13]=f2bf(h3.y); A[14]=f2bf(h3.z); A[15]=f2bf(h3.w);
        // B (32x16 bf16): K = half*8 + {0..7} and {16..23}, N = m
        const float* wcol = W + (size_t)(kt + half * 8) * GAT_OUTF + n0 + m;
        #pragma unroll
        for (int v = 0; v < 8; ++v) {
            B[2*v]   = f2bf(wcol[(2*v + (v >= 4 ? 8 : 0))     * GAT_OUTF]);
            B[2*v+1] = f2bf(wcol[(2*v + (v >= 4 ? 8 : 0) + 1) * GAT_OUTF]);
        }
        acc = __builtin_amdgcn_wmma_f32_16x16x32_bf16(false, A, false, B,
                                                      (short)0, acc, false, false);
    }
    // D: lane holds rows r0 + 8*half + {0..7} at column n0+m -> contiguous
    // in the transposed layouts.
    v8bf bacc;
    #pragma unroll
    for (int r = 0; r < 8; ++r) bacc[r] = f2bf(acc[r]);
    const size_t base = (size_t)(n0 + m) * GAT_N + r0 + 8 * half;
    *(v8f*) (WhfT + base) = acc;
    *(v8bf*)(WhbT + base) = bacc;
}

// -------------------------------------------------------------------------
// Kernel 2: src = Wh@a1, dst = Wh@a2 from the transposed WhfT.
// One thread per row; loads coalesce across threads, a[] folds to s_loads.
// -------------------------------------------------------------------------
__global__ void gat_srcdst_kernel(const float* __restrict__ WhfT,
                                  const float* __restrict__ a,
                                  float* __restrict__ src,
                                  float* __restrict__ dst)
{
    const int row = blockIdx.x * 256 + threadIdx.x;
    float s = 0.f, d = 0.f;
    #pragma unroll 4
    for (int n = 0; n < GAT_OUTF; ++n) {
        const float w = WhfT[(size_t)n * GAT_N + row];
        s += w * a[n];
        d += w * a[GAT_OUTF + n];
    }
    src[row] = s;
    dst[row] = d;
}

// -------------------------------------------------------------------------
// Kernel 3: dmax = max(dst). Single block.
// -------------------------------------------------------------------------
__global__ void gat_dstmax_kernel(const float* __restrict__ dst,
                                  float* __restrict__ dmax)
{
    __shared__ float red[256];
    float mx = -3.0e38f;
    for (int i = threadIdx.x; i < GAT_N; i += 256) mx = fmaxf(mx, dst[i]);
    red[threadIdx.x] = mx;
    __syncthreads();
    for (int s = 128; s > 0; s >>= 1) {
        if (threadIdx.x < s) red[threadIdx.x] = fmaxf(red[threadIdx.x], red[threadIdx.x + s]);
        __syncthreads();
    }
    if (threadIdx.x == 0) dmax[0] = red[0];
}

// -------------------------------------------------------------------------
// Kernel 4: fused masked-softmax attention + P@Wh + elu.
// Block = 8 waves sharing one 16-row tile; each wave scans 1024 columns.
// Stabilizer M_i = lrelu(src_i + max(dst)) is an exact upper bound (lrelu
// monotone), so partial (l, acc) over column chunks combine by addition.
// P built directly in WMMA A bf16 layout; B fragments are contiguous 32-byte
// loads from the transposed WhbT. dst is staged in LDS; the 32 KB region is
// reused (union) for the cross-wave accumulator reduction afterwards.
// -------------------------------------------------------------------------
union GatShMem {
    float dst[GAT_N];        // 32 KB, live during the main loop
    float acc[4 * 2048];     // 32 KB, live during the reduction
};

__global__ void gat_attn_kernel(const int* __restrict__ adj,
                                const float* __restrict__ srcA,
                                const float* __restrict__ dstA,
                                const float* __restrict__ dmaxp,
                                const __bf16* __restrict__ WhbT,
                                float* __restrict__ out)
{
    __shared__ GatShMem sh;
    __shared__ float sL[8 * 16];
    __shared__ float sLtot[16];

    const int lane = threadIdx.x & 31;
    const int wave = threadIdx.x >> 5;
    const int half = lane >> 4;
    const int m    = lane & 15;
    const int r0   = blockIdx.x * 16;
    const int row  = r0 + m;

    // stage dst[] into LDS with b128 loads
    {
        const float4* dp = (const float4*)dstA;
        float4* sp = (float4*)sh.dst;
        #pragma unroll
        for (int i = threadIdx.x; i < GAT_N / 4; i += 256) sp[i] = dp[i];
    }

    const float srcv = srcA[row];
    const float dmax = dmaxp[0];
    float rowM = srcv + dmax;
    rowM = rowM > 0.f ? rowM : GAT_ALPHA * rowM;

    const int* adjrow = adj + (size_t)row * GAT_N;
    float lsum = 0.f;
    v8f acc[8] = {};

    __syncthreads();

    const int c0 = wave * (GAT_N / 8);
    for (int ct = c0; ct < c0 + GAT_N / 8; ct += 32) {
        __builtin_prefetch(adjrow + ct + 512, 0, 1);
        v16bf P;
        #pragma unroll
        for (int v = 0; v < 8; ++v) {
            const int kb  = 2*v + half*16;
            const int col = ct + kb;
            const int2   aa = *(const int2*)  (adjrow + col);
            const float2 dd = *(const float2*)(sh.dst + col);
            float e0 = srcv + dd.x; e0 = e0 > 0.f ? e0 : GAT_ALPHA * e0;
            float e1 = srcv + dd.y; e1 = e1 > 0.f ? e1 : GAT_ALPHA * e1;
            const float p0 = aa.x ? __expf(e0 - rowM) : 0.f;
            const float p1 = aa.y ? __expf(e1 - rowM) : 0.f;
            lsum += p0 + p1;
            P[2*v]   = f2bf(p0);
            P[2*v+1] = f2bf(p1);
        }
        #pragma unroll
        for (int n = 0; n < 8; ++n) {
            // B (32x16): lane m = column n*16+m, K = ct + half*16 + {0..15}
            // contiguous in WhbT -> one 32-byte fragment load.
            const v16bf B = *(const v16bf*)(WhbT + (size_t)(n * 16 + m) * GAT_N
                                                  + ct + half * 16);
            acc[n] = __builtin_amdgcn_wmma_f32_16x16x32_bf16(false, P, false, B,
                                                             (short)0, acc[n], false, false);
        }
    }

    // combine the two K-halves of l per row
    lsum += __shfl_xor(lsum, 16, 32);
    if (half == 0) sL[wave * 16 + m] = lsum;

    __syncthreads();   // dst staging dead; LDS region becomes the accumulator

    // two-stage LDS reduction of the 16x128 f32 partials across 8 waves
    float* sa = sh.acc + (wave & 3) * 2048;
    if (wave >= 4) {
        #pragma unroll
        for (int n = 0; n < 8; ++n)
            #pragma unroll
            for (int r = 0; r < 8; ++r)
                sa[(r + 8 * half) * GAT_OUTF + n * 16 + m] = acc[n][r];
    }
    __syncthreads();
    if (wave < 4) {
        #pragma unroll
        for (int n = 0; n < 8; ++n)
            #pragma unroll
            for (int r = 0; r < 8; ++r)
                sa[(r + 8 * half) * GAT_OUTF + n * 16 + m] += acc[n][r];
    }
    if (threadIdx.x < 16) {
        float t = 0.f;
        #pragma unroll
        for (int w = 0; w < 8; ++w) t += sL[w * 16 + threadIdx.x];
        sLtot[threadIdx.x] = t;
    }
    __syncthreads();

    // epilogue: sum 4 partials, divide by softmax denom, elu, store
    for (int i = threadIdx.x; i < 2048; i += 256) {
        const float s  = sh.acc[i] + sh.acc[2048 + i] + sh.acc[4096 + i] + sh.acc[6144 + i];
        const int   lr = i >> 7;
        const float x  = s / sLtot[lr];
        out[(size_t)(r0 + lr) * GAT_OUTF + (i & 127)] = x > 0.f ? x : (__expf(x) - 1.f);
    }
}

// -------------------------------------------------------------------------
extern "C" void kernel_launch(void* const* d_in, const int* in_sizes, int n_in,
                              void* d_out, int out_size, void* d_ws, size_t ws_size,
                              hipStream_t stream)
{
    const float* h   = (const float*)d_in[0];
    const int*   adj = (const int*)  d_in[1];
    const float* W   = (const float*)d_in[2];
    const float* a   = (const float*)d_in[3];
    float* out = (float*)d_out;

    char* ws = (char*)d_ws;
    float*  WhfT = (float*) (ws);              // 128*8192 f32  = 4 MB (transposed)
    __bf16* WhbT = (__bf16*)(ws + 4194304);    // 128*8192 bf16 = 2 MB (transposed)
    float*  src  = (float*) (ws + 6291456);    // 32 KB
    float*  dst  = (float*) (ws + 6324224);    // 32 KB
    float*  dmax = (float*) (ws + 6356992);    // 4 B

    gat_wh_kernel    <<<512, 256, 0, stream>>>(h, W, WhfT, WhbT);
    gat_srcdst_kernel<<<32, 256, 0, stream>>>(WhfT, a, src, dst);
    gat_dstmax_kernel<<<1, 256, 0, stream>>>(dst, dmax);
    gat_attn_kernel  <<<512, 256, 0, stream>>>(adj, src, dst, dmax, WhbT, out);
}
